// PFNLayerV2_16174846837184
// MI455X (gfx1250) — matrix operand order repaired and checked by reading, not compiled, not run on hardware
//
#include <hip/hip_runtime.h>

#define N_PTS  1000000
#define C_IN   10
#define C_OUT  64
#define SEGS   30000
#define BN_EPS 1e-3f

#define TILE        512     // points per block
#define THREADS     256     // 8 waves (wave32)
#define MT_PER_WAVE 4       // 16-point tiles per wave

// workspace layout (float offsets)
#define WS_SUM1   0
#define WS_SUM2   64
#define WS_BNA    128
#define WS_BNB    192
#define WS_GSUM   256
#define WS_XG     320
#define WS_SEGMAX 512       // SEGS*64 floats

typedef __attribute__((ext_vector_type(16))) _Float16 v16h;
typedef __attribute__((ext_vector_type(8)))  _Float16 v8h;
typedef __attribute__((ext_vector_type(8)))  float    v8f;

union Frag16 { v16h v; v8h h[2]; };

// A-fragment (16x32 f16, M x K): lanes 0-15 hold M=0..15 / K=base0..base0+7 and 16+base0..,
// lanes 16-31 have base0 = 8.  Two 16-byte LDS loads.
__device__ __forceinline__ v16h load_a_frag(const _Float16* base, int row0, int ldk, int kc, int lane) {
  const int m  = lane & 15;
  const int b0 = (lane >> 4) << 3;
  const _Float16* p = base + (row0 + m) * ldk + kc + b0;
  Frag16 u;
  u.h[0] = *(const v8h*)(p);
  u.h[1] = *(const v8h*)(p + 16);
  return u.v;
}

// B-fragment (32x16 f16, K x N): lane col n = lane&15, K = kc + 8*(lane>=16) + j (j<8), kc + 16 + 8*(lane>=16) + (j-8) (j>=8).
// B[K][o] = W[o][K] -> two 16-byte row chunks (k and k+16) per lane.
__device__ __forceinline__ v16h load_b_frag(const _Float16* w, int ntile, int ldk, int kc, int lane) {
  const int o  = ntile * 16 + (lane & 15);
  const int kb = kc + ((lane >> 4) << 3);
  const _Float16* p = w + o * ldk + kb;
  Frag16 u;
  u.h[0] = *(const v8h*)(p);
  u.h[1] = *(const v8h*)(p + 16);
  return u.v;
}

__device__ __forceinline__ float sigmoidf_(float x) { return 1.f / (1.f + __expf(-x)); }

__device__ __forceinline__ void flush_seg(float* out_sum, float* segmax, int seg, int ch,
                                          float runs, float runm) {
  if (seg >= 0) {
    atomicAdd(&out_sum[seg * C_OUT + ch], runs);
    atomicMax((unsigned int*)&segmax[seg * C_OUT + ch], __float_as_uint(runm)); // xi >= 0
  }
}

// ---------------------------------------------------------------- init (runs every launch)
__global__ void k_init(float* __restrict__ out, float* __restrict__ ws, int n_out, int n_ws) {
  int i = blockIdx.x * blockDim.x + threadIdx.x;
  int stride = gridDim.x * blockDim.x;
  for (int j = i; j < n_out; j += stride) out[j] = 0.f;
  for (int j = i; j < n_ws;  j += stride) ws[j]  = 0.f;
}

// ---------------------------------------------------------------- pass 1: moments of x = inp @ Wlin^T
__global__ __launch_bounds__(THREADS) void k_linstats(const float* __restrict__ inp,
                                                      const float* __restrict__ Wlin,
                                                      float* __restrict__ ws) {
  __shared__ __align__(16) _Float16 a_lds[TILE * 32];
  __shared__ __align__(16) _Float16 wl[C_OUT * 32];
  __shared__ float red[128];
  const int tid  = threadIdx.x;
  const int base = blockIdx.x * TILE;

  for (int idx = tid; idx < C_OUT * 32; idx += THREADS) {
    int o = idx >> 5, k = idx & 31;
    wl[idx] = (_Float16)(k < C_IN ? Wlin[o * C_IN + k] : 0.f);
  }
  for (int idx = tid; idx < TILE * 32; idx += THREADS) {
    int pt = idx >> 5, k = idx & 31;
    int p = base + pt;
    float v = (k < C_IN && p < N_PTS) ? inp[p * C_IN + k] : 0.f;   // padded rows -> x==0, harmless
    a_lds[idx] = (_Float16)v;
  }
  if (tid < 128) red[tid] = 0.f;
  __syncthreads();

  const int lane = tid & 31, w = tid >> 5;
  float s1[4] = {0.f,0.f,0.f,0.f}, s2[4] = {0.f,0.f,0.f,0.f};
  for (int rep = 0; rep < MT_PER_WAVE; ++rep) {
    int r0 = (w * MT_PER_WAVE + rep) * 16;
    v16h a = load_a_frag(a_lds, r0, 32, 0, lane);
#pragma unroll
    for (int t = 0; t < 4; ++t) {
      v16h b = load_b_frag(wl, t, 32, 0, lane);
      v8f acc = {};
      acc = __builtin_amdgcn_wmma_f32_16x16x32_f16(false, a, false, b, (short)0, acc, false, false);
#pragma unroll
      for (int i = 0; i < 8; ++i) { float x = acc[i]; s1[t] += x; s2[t] += x * x; }
    }
  }
#pragma unroll
  for (int t = 0; t < 4; ++t) {
    int ch = t * 16 + (lane & 15);
    atomicAdd(&red[ch], s1[t]);
    atomicAdd(&red[64 + ch], s2[t]);
  }
  __syncthreads();
  if (tid < 64) {
    atomicAdd(&ws[WS_SUM1 + tid], red[tid]);
    atomicAdd(&ws[WS_SUM2 + tid], red[64 + tid]);
  }
}

// ---------------------------------------------------------------- BN scale/shift
__global__ void k_bnparams(const float* __restrict__ gamma, const float* __restrict__ beta,
                           float* __restrict__ ws) {
  int c = threadIdx.x;
  if (c < C_OUT) {
    float inv_n = 1.f / (float)N_PTS;
    float mu  = ws[WS_SUM1 + c] * inv_n;
    float var = ws[WS_SUM2 + c] * inv_n - mu * mu;
    float a = gamma[c] * rsqrtf(var + BN_EPS);
    ws[WS_BNA + c] = a;
    ws[WS_BNB + c] = beta[c] - mu * a;
  }
}

// ---------------------------------------------------------------- pass 2: g = mean(relu(bn(x)))
__global__ __launch_bounds__(THREADS) void k_gmean(const float* __restrict__ inp,
                                                   const float* __restrict__ Wlin,
                                                   float* __restrict__ ws) {
  __shared__ __align__(16) _Float16 a_lds[TILE * 32];
  __shared__ __align__(16) _Float16 wl[C_OUT * 32];
  __shared__ float red[64];
  const int tid  = threadIdx.x;
  const int base = blockIdx.x * TILE;

  for (int idx = tid; idx < C_OUT * 32; idx += THREADS) {
    int o = idx >> 5, k = idx & 31;
    wl[idx] = (_Float16)(k < C_IN ? Wlin[o * C_IN + k] : 0.f);
  }
  for (int idx = tid; idx < TILE * 32; idx += THREADS) {
    int pt = idx >> 5, k = idx & 31;
    int p = base + pt;
    float v = (k < C_IN && p < N_PTS) ? inp[p * C_IN + k] : 0.f;
    a_lds[idx] = (_Float16)v;
  }
  if (tid < 64) red[tid] = 0.f;
  __syncthreads();

  const int lane = tid & 31, w = tid >> 5;
  float bna[4], bnb[4];
#pragma unroll
  for (int t = 0; t < 4; ++t) {
    int ch = t * 16 + (lane & 15);
    bna[t] = ws[WS_BNA + ch];
    bnb[t] = ws[WS_BNB + ch];
  }
  float gs[4] = {0.f,0.f,0.f,0.f};
  for (int rep = 0; rep < MT_PER_WAVE; ++rep) {
    int r0 = (w * MT_PER_WAVE + rep) * 16;
    v16h a = load_a_frag(a_lds, r0, 32, 0, lane);
#pragma unroll
    for (int t = 0; t < 4; ++t) {
      v16h b = load_b_frag(wl, t, 32, 0, lane);
      v8f acc = {};
      acc = __builtin_amdgcn_wmma_f32_16x16x32_f16(false, a, false, b, (short)0, acc, false, false);
#pragma unroll
      for (int i = 0; i < 8; ++i) {
        int m = i + ((lane >> 4) << 3);
        if (base + r0 + m < N_PTS) {                 // padded rows would contribute relu(bnb)
          float v = fmaxf(acc[i] * bna[t] + bnb[t], 0.f);
          gs[t] += v;
        }
      }
    }
  }
#pragma unroll
  for (int t = 0; t < 4; ++t) atomicAdd(&red[t * 16 + (lane & 15)], gs[t]);
  __syncthreads();
  if (tid < 64) atomicAdd(&ws[WS_GSUM + tid], red[tid]);
}

// ---------------------------------------------------------------- global branch: xg = pw2 @ relu(dw2*g)
__global__ void k_xg(const float* __restrict__ dw2, const float* __restrict__ pw2,
                     float* __restrict__ ws) {
  __shared__ float gact[C_OUT];
  int t = threadIdx.x;
  if (t < C_OUT) {
    float g = ws[WS_GSUM + t] / (float)N_PTS;
    float s = dw2[t] * g;
    gact[t] = s > 0.f ? s : 0.f;
  }
  __syncthreads();
  if (t < C_OUT) {
    float acc = 0.f;
#pragma unroll 8
    for (int k = 0; k < C_OUT; ++k) acc += pw2[t * C_OUT + k] * gact[k];
    ws[WS_XG + t] = acc;
  }
}

// ---------------------------------------------------------------- pass 3: fused main + segment reduce
__global__ __launch_bounds__(THREADS) void k_main(const float* __restrict__ inp,
                                                  const int*   __restrict__ unq,
                                                  const float* __restrict__ Wlin,
                                                  const float* __restrict__ dw1,
                                                  const float* __restrict__ pw1,
                                                  float* __restrict__ ws,
                                                  float* __restrict__ out) {
  __shared__ __align__(16) _Float16 a_lds[TILE * 32];      // padded f16 inputs
  __shared__ __align__(16) _Float16 wl[C_OUT * 32];        // Wlin f16 (K padded to 32)
  __shared__ __align__(16) _Float16 pwh[C_OUT * C_OUT];    // pw1 f16
  __shared__ __align__(16) _Float16 s_lds[8][16 * C_OUT];  // per-wave swish tile
  __shared__ int iv[TILE];                                 // pillar ids
  const int tid  = threadIdx.x;
  const int base = blockIdx.x * TILE;
  float* segmax = ws + WS_SEGMAX;

  if (base + TILE < N_PTS) __builtin_prefetch(inp + (base + TILE) * C_IN + tid, 0, 1);

  for (int idx = tid; idx < C_OUT * 32; idx += THREADS) {
    int o = idx >> 5, k = idx & 31;
    wl[idx] = (_Float16)(k < C_IN ? Wlin[o * C_IN + k] : 0.f);
  }
  for (int idx = tid; idx < C_OUT * C_OUT; idx += THREADS)
    pwh[idx] = (_Float16)pw1[idx];
  for (int idx = tid; idx < TILE * 32; idx += THREADS) {
    int pt = idx >> 5, k = idx & 31;
    int p = base + pt;
    float v = (k < C_IN && p < N_PTS) ? inp[p * C_IN + k] : 0.f;
    a_lds[idx] = (_Float16)v;
  }
  for (int idx = tid; idx < TILE; idx += THREADS) {
    int p = base + idx;
    iv[idx] = (p < N_PTS) ? unq[p] : -1;
  }
  __syncthreads();

  const int lane = tid & 31, w = tid >> 5;
  float bna[4], bnb[4], ldw[4], lxg[4];
#pragma unroll
  for (int t = 0; t < 4; ++t) {
    int ch = t * 16 + (lane & 15);
    bna[t] = ws[WS_BNA + ch];
    bnb[t] = ws[WS_BNB + ch];
    ldw[t] = dw1[ch];
    lxg[t] = ws[WS_XG + ch];
  }
  _Float16* sw = s_lds[w];

  for (int rep = 0; rep < MT_PER_WAVE; ++rep) {
    const int r0 = (w * MT_PER_WAVE + rep) * 16;

    // ---- linear (K padded 10->32): 4 WMMAs
    v16h a = load_a_frag(a_lds, r0, 32, 0, lane);
    v8f x[4];
#pragma unroll
    for (int t = 0; t < 4; ++t) {
      v16h b = load_b_frag(wl, t, 32, 0, lane);
      v8f acc = {};
      x[t] = __builtin_amdgcn_wmma_f32_16x16x32_f16(false, a, false, b, (short)0, acc, false, false);
    }

    // ---- BN + ReLU, keep x in regs; swish(x*dw1) -> per-wave LDS tile (f16)
#pragma unroll
    for (int t = 0; t < 4; ++t) {
      int ch = t * 16 + (lane & 15);
#pragma unroll
      for (int i = 0; i < 8; ++i) {
        float v = fmaxf(x[t][i] * bna[t] + bnb[t], 0.f);
        x[t][i] = v;
        float s = v * ldw[t];
        float act = s * sigmoidf_(s);                 // swish
        int m = i + ((lane >> 4) << 3);
        sw[m * C_OUT + ch] = (_Float16)act;
      }
    }
    // DS ops are in-order per wave: drain stores so cross-lane fragment loads see them
    asm volatile("s_wait_dscnt 0" ::: "memory");

    // ---- pointwise conv: xl = swish_tile @ pw1^T, K=64 -> 8 WMMAs
    v8f xl[4] = {};
#pragma unroll
    for (int kc = 0; kc < C_OUT; kc += 32) {
      v16h a2 = load_a_frag(sw, 0, C_OUT, kc, lane);
#pragma unroll
      for (int t = 0; t < 4; ++t) {
        v16h b2 = load_b_frag(pwh, t, C_OUT, kc, lane);
        xl[t] = __builtin_amdgcn_wmma_f32_16x16x32_f16(false, a2, false, b2, (short)0, xl[t], false, false);
      }
    }

    // ---- gate + run-length segment reduce (each lane: 8 consecutive points, 1 channel)
#pragma unroll
    for (int t = 0; t < 4; ++t) {
      int ch = t * 16 + (lane & 15);
      float runs = 0.f, runm = 0.f;
      int cur = -1;
#pragma unroll
      for (int i = 0; i < 8; ++i) {
        int m = i + ((lane >> 4) << 3);
        float wei = sigmoidf_(xl[t][i] + lxg[t]);
        float xi  = x[t][i] * (1.f + wei);            // >= 0
        int seg = iv[r0 + m];                          // -1 for padded rows
        if (seg != cur) {
          flush_seg(out, segmax, cur, ch, runs, runm);
          cur = seg; runs = xi; runm = xi;
        } else {
          runs += xi; runm = fmaxf(runm, xi);
        }
      }
      flush_seg(out, segmax, cur, ch, runs, runm);
    }
  }
}

// ---------------------------------------------------------------- out = seg_sum (already in out) + seg_max
__global__ void k_final(float* __restrict__ out, const float* __restrict__ segmax) {
  int i = blockIdx.x * blockDim.x + threadIdx.x;
  if (i < SEGS * C_OUT) out[i] += segmax[i];
}

extern "C" void kernel_launch(void* const* d_in, const int* in_sizes, int n_in,
                              void* d_out, int out_size, void* d_ws, size_t ws_size,
                              hipStream_t stream) {
  (void)in_sizes; (void)n_in; (void)out_size; (void)ws_size;
  const float* inp   = (const float*)d_in[0];
  const int*   unq   = (const int*)  d_in[1];
  const float* Wlin  = (const float*)d_in[2];
  const float* gamma = (const float*)d_in[3];
  const float* beta  = (const float*)d_in[4];
  const float* dw1   = (const float*)d_in[5];
  const float* pw2a  = (const float*)d_in[6];   // pw1
  const float* dw2   = (const float*)d_in[7];
  const float* pw2   = (const float*)d_in[8];
  float* out = (float*)d_out;
  float* ws  = (float*)d_ws;

  const int nblk = (N_PTS + TILE - 1) / TILE;
  const int n_out = SEGS * C_OUT;
  const int n_ws  = WS_SEGMAX + SEGS * C_OUT;

  k_init    <<<2048, 256, 0, stream>>>(out, ws, n_out, n_ws);
  k_linstats<<<nblk, THREADS, 0, stream>>>(inp, Wlin, ws);
  k_bnparams<<<1, 64, 0, stream>>>(gamma, beta, ws);
  k_gmean   <<<nblk, THREADS, 0, stream>>>(inp, Wlin, ws);
  k_xg      <<<1, 64, 0, stream>>>(dw2, pw2, ws);
  k_main    <<<nblk, THREADS, 0, stream>>>(inp, unq, Wlin, dw1, pw2a, ws, out);
  k_final   <<<(n_out + 255) / 256, 256, 0, stream>>>(out, ws + WS_SEGMAX);
}
